// FPSTokenizer_23373212025160
// MI455X (gfx1250) — matrix operand order repaired and checked by reading, not compiled, not run on hardware
//
#include <hip/hip_runtime.h>
#include <cstdint>
#include <cstddef>

// ---------------------------------------------------------------------------
// MI455X (gfx1250) implementation: bf16 WMMA GEMMs for MLP1/MLP2 with a
// double-buffered GLOBAL_LOAD_ASYNC_TO_LDS pipeline (ASYNCcnt-tracked),
// LDS-resident FPS + kNN selection. Wave32, WMMA 16x16x32 bf16, f32 accum.
// ---------------------------------------------------------------------------

typedef __attribute__((ext_vector_type(16))) __bf16 v16bf;
typedef __attribute__((ext_vector_type(8)))  float  v8f;

#define NTOT    131072
#define BATCH   8
#define NPTS    16384
#define FEAT    32
#define TOKDIM  768
#define MAXTOK  128
#define KNN_K   16

// ----- CDNA5 async global->LDS path (guarded: falls back to sync copy) -----
#if __has_builtin(__builtin_amdgcn_global_load_async_to_lds_b128)
#define ASYNC_LDS 1
// Builtin signature (from clang diagnostic): first arg is a pointer to a
// 16-byte int vector in the global AS, second in the LDS AS.
typedef int v4i __attribute__((vector_size(16)));
typedef __attribute__((address_space(1))) v4i* gas_v4i_p;
typedef __attribute__((address_space(3))) v4i* las_v4i_p;
#else
#define ASYNC_LDS 0
#endif

template <int N>
__device__ __forceinline__ void wait_async_le() {
#if __has_builtin(__builtin_amdgcn_s_wait_asynccnt)
    __builtin_amdgcn_s_wait_asynccnt(N);
#else
    asm volatile("s_wait_asynccnt %0" :: "i"(N) : "memory");
#endif
}

__device__ __forceinline__ float gelu_exact(float x) {
    // torch nn.GELU default: 0.5*x*(1+erf(x/sqrt(2)))
    return 0.5f * x * (1.0f + erff(x * 0.70710678118654752440f));
}

// ---------------------------------------------------------------------------
// Tiled bf16 GEMM: C[M,N] = act(A[M,K] @ B[K,N] + bias), B given pre-transposed
// as Bt[N,K]. Block = 256 threads (8 waves), tile 128(M) x 128(N), k-step 32,
// double-buffered async global->LDS staging. Each wave computes a 32x64
// sub-tile = 8 v_wmma_f32_16x16x32_bf16 per k-step.
// Requires M%128==0, N%128==0, K%32==0 (true for all calls here).
// ---------------------------------------------------------------------------
union FragU { v16bf v; uint4 q[2]; };

__global__ __launch_bounds__(256) void gemm_bf16_wmma(
    const __bf16* __restrict__ A,    // [M,K] row-major bf16
    const __bf16* __restrict__ Bt,   // [N,K] row-major bf16 (transposed weights)
    const float*  __restrict__ bias, // [N] f32
    void* __restrict__ out,          // bf16 [M,N] or f32 [M,N]
    int M, int N, int K, int act, int out_f32)
{
    // Double-buffered LDS tiles: [128 rows][32 bf16], row stride padded to 40
    // bf16 (80B) so the 16B fragment reads across lanes hit distinct banks.
    __shared__ __bf16 sA[2][128 * 40];
    __shared__ __bf16 sB[2][128 * 40];

    const int t    = threadIdx.x;
    const int lane = t & 31;
    const int w    = t >> 5;
    const int wm   = w & 3;        // 4 waves along M, 32 rows each
    const int wn   = w >> 2;       // 2 waves along N, 64 cols each
    const int blkN = blockIdx.x * 128;
    const int blkM = blockIdx.y * 128;

    v8f acc[2][4];
#pragma unroll
    for (int i = 0; i < 2; ++i)
#pragma unroll
        for (int j = 0; j < 4; ++j)
#pragma unroll
            for (int r = 0; r < 8; ++r) acc[i][j][r] = 0.0f;

    // Cooperative tile-copy mapping: 256 threads x 32B = 8192B per tile.
    const int lr = t >> 1;   // row 0..127 within tile
    const int lh = t & 1;    // which 32-byte half of the 64-byte row

    // Issue one k-tile (A + Bt) into LDS buffer `buf`:
    // 4 x b128 async copies per thread (2 for A, 2 for Bt) -> ASYNCcnt += 4.
    auto issue_tile = [&](int kt, int buf) {
        const __bf16* ga = A  + (size_t)(blkM + lr) * (size_t)K + kt + lh * 16;
        const __bf16* gb = Bt + (size_t)(blkN + lr) * (size_t)K + kt + lh * 16;
        __bf16* la = &sA[buf][lr * 40 + lh * 16];
        __bf16* lb = &sB[buf][lr * 40 + lh * 16];
#if ASYNC_LDS
        __builtin_amdgcn_global_load_async_to_lds_b128((gas_v4i_p)ga,       (las_v4i_p)la,       0, 0);
        __builtin_amdgcn_global_load_async_to_lds_b128((gas_v4i_p)(ga + 8), (las_v4i_p)(la + 8), 0, 0);
        __builtin_amdgcn_global_load_async_to_lds_b128((gas_v4i_p)gb,       (las_v4i_p)lb,       0, 0);
        __builtin_amdgcn_global_load_async_to_lds_b128((gas_v4i_p)(gb + 8), (las_v4i_p)(lb + 8), 0, 0);
#else
        const uint4* pga = reinterpret_cast<const uint4*>(ga);
        uint4 a0 = pga[0], a1 = pga[1];
        const uint4* pgb = reinterpret_cast<const uint4*>(gb);
        uint4 b0 = pgb[0], b1 = pgb[1];
        uint4* pla = reinterpret_cast<uint4*>(la);
        pla[0] = a0; pla[1] = a1;
        uint4* plb = reinterpret_cast<uint4*>(lb);
        plb[0] = b0; plb[1] = b1;
#endif
    };

    const int nk = K >> 5;
    issue_tile(0, 0);

    for (int ki = 0; ki < nk; ++ki) {
        const int cur = ki & 1;
        const bool more = (ki + 1 < nk);
        if (more) issue_tile((ki + 1) << 5, cur ^ 1);   // prefetch next tile
#if ASYNC_LDS
        // Older 4 copies (buffer `cur`) complete; the 4 just issued may remain
        // in flight behind the WMMAs of this iteration.
        if (more) wait_async_le<4>(); else wait_async_le<0>();
#endif
        __syncthreads();   // buffer `cur` visible to all waves

        // Fragment loads per ISA 7.12.2 (wave32 bf16 16x16x32):
        //  A: lane L -> m=L&15; k = (j&7) + 16*(j>>3) + 8*(L>>4)   (2x16B reads)
        //  B: lane L -> n=L&15; k = (j&7) + 16*(j>>3) + 8*(L>>4)   (2x16B reads)
        v16bf af[2], bfrag[4];
        {
            const int mn = lane & 15, hi = lane >> 4;
#pragma unroll
            for (int i = 0; i < 2; ++i) {
                const __bf16* base = &sA[cur][(wm * 32 + i * 16 + mn) * 40 + hi * 8];
                FragU f;
                f.q[0] = *reinterpret_cast<const uint4*>(base);        // k: hi*8 .. +7
                f.q[1] = *reinterpret_cast<const uint4*>(base + 16);   // k: 16+hi*8 ..
                af[i] = f.v;
            }
#pragma unroll
            for (int j = 0; j < 4; ++j) {
                const __bf16* base = &sB[cur][(wn * 64 + j * 16 + mn) * 40 + hi * 8];
                FragU f;
                f.q[0] = *reinterpret_cast<const uint4*>(base);        // k: hi*8 .. +7
                f.q[1] = *reinterpret_cast<const uint4*>(base + 16);   // k: 16+hi*8 ..
                bfrag[j] = f.v;
            }
        }
#pragma unroll
        for (int i = 0; i < 2; ++i)
#pragma unroll
            for (int j = 0; j < 4; ++j)
                acc[i][j] = __builtin_amdgcn_wmma_f32_16x16x32_bf16(
                    false, af[i], false, bfrag[j], (short)0, acc[i][j], false, false);

        __syncthreads();   // all reads of buffer `cur` done before it is reused
    }

    // Epilogue: C/D layout lane L -> col=L&15, row = r + 8*(L>>4).
    const int cn = lane & 15;
    const int rh = (lane >> 4) * 8;
#pragma unroll
    for (int j = 0; j < 4; ++j) {
        const int col = blkN + wn * 64 + j * 16 + cn;
        const float bv = bias[col];
#pragma unroll
        for (int i = 0; i < 2; ++i) {
#pragma unroll
            for (int r = 0; r < 8; ++r) {
                const int row = blkM + wm * 32 + i * 16 + rh + r;
                float x = acc[i][j][r] + bv;
                if (act) x = gelu_exact(x);
                const size_t idx = (size_t)row * (size_t)N + col;
                if (out_f32) ((float*)out)[idx] = x;
                else         ((__bf16*)out)[idx] = (__bf16)x;
            }
        }
    }
}

// ---------------------------------------------------------------------------
// Small prep kernels
// ---------------------------------------------------------------------------
__global__ void pack_points(const float* __restrict__ coords,
                            const float* __restrict__ times,
                            float4* __restrict__ P4, int n) {
    int i = blockIdx.x * blockDim.x + threadIdx.x;
    if (i < n) P4[i] = make_float4(coords[3*i], coords[3*i+1], coords[3*i+2], times[i]);
}

__global__ void cvt_f32_bf16(const float* __restrict__ in, __bf16* __restrict__ out, int n) {
    int i = blockIdx.x * blockDim.x + threadIdx.x;
    if (i < n) out[i] = (__bf16)in[i];
}

// W[K,N] f32 -> Wt[N,K] bf16
__global__ void cvt_transpose_w(const float* __restrict__ W, __bf16* __restrict__ Wt,
                                int K, int N) {
    int i = blockIdx.x * blockDim.x + threadIdx.x;
    if (i < K * N) {
        int k = i / N, n = i % N;
        Wt[(size_t)n * K + k] = (__bf16)W[i];
    }
}

__global__ void fill_ones(float* __restrict__ p, int n) {
    int i = blockIdx.x * blockDim.x + threadIdx.x;
    if (i < n) p[i] = 1.0f;
}

// ---------------------------------------------------------------------------
// FPS: one 1024-thread WG per batch. min_d[16384] lives in LDS (64KB of the
// 320KB WGP LDS). 128 sequential steps, each: min-distance update + argmax
// tree reduction. Tie-breaking matches jnp.argmax (first/lowest index).
// ---------------------------------------------------------------------------
__global__ __launch_bounds__(1024) void fps_kernel(const float4* __restrict__ P4,
                                                   int*   __restrict__ fps_idx,
                                                   float* __restrict__ cents_ws,
                                                   float* __restrict__ cents_out) {
    __shared__ float mind[NPTS];
    __shared__ float rv[1024];
    __shared__ int   ri[1024];
    __shared__ int   s_last;
    const int b = blockIdx.x, t = threadIdx.x;
    const float4* Pb = P4 + (size_t)b * NPTS;

    for (int i = t; i < NPTS; i += 1024) mind[i] = 3.4e38f;
    if (t == 0) s_last = 0;
    __syncthreads();

    for (int s = 0; s < MAXTOK; ++s) {
        const int last = s_last;
        const float4 pt = Pb[last];
        if (t == 0) {
            fps_idx[b * MAXTOK + s] = last;
            float* cw = cents_ws  + (size_t)(b * MAXTOK + s) * 4;
            float* co = cents_out + (size_t)(b * MAXTOK + s) * 4;
            cw[0] = pt.x; cw[1] = pt.y; cw[2] = pt.z; cw[3] = pt.w;
            co[0] = pt.x; co[1] = pt.y; co[2] = pt.z; co[3] = pt.w;
        }
        float best = -1.0f; int bi = 0;
        for (int i = t; i < NPTS; i += 1024) {
            float4 q = Pb[i];
            float dx = q.x - pt.x, dy = q.y - pt.y, dz = q.z - pt.z, dw = q.w - pt.w;
            float d  = dx*dx + dy*dy + dz*dz + dw*dw;
            float md = fminf(mind[i], d);
            mind[i] = md;
            if (md > best) { best = md; bi = i; }   // ascending i: keeps first max
        }
        rv[t] = best; ri[t] = bi;
        __syncthreads();
        for (int off = 512; off > 0; off >>= 1) {
            if (t < off) {
                float v2 = rv[t + off]; int i2 = ri[t + off];
                if (v2 > rv[t] || (v2 == rv[t] && i2 < ri[t])) { rv[t] = v2; ri[t] = i2; }
            }
            __syncthreads();
        }
        if (t == 0) s_last = ri[0];
        __syncthreads();
    }
}

// ---------------------------------------------------------------------------
// kNN (top-16 by squared distance, tie -> lower index, matching top_k) fused
// with feature gather + max-pool. One 256-thread WG per (batch, centroid).
// Distances live in LDS (64KB); 16 iterative argmin reductions; then pool 768
// bf16 feature dims over the 16 selected rows.
// ---------------------------------------------------------------------------
__global__ __launch_bounds__(256) void knn_pool_kernel(const float4* __restrict__ P4,
                                                       const float*  __restrict__ cents,
                                                       const __bf16* __restrict__ feats,
                                                       __bf16* __restrict__ pooled) {
    __shared__ float dd[NPTS];
    __shared__ float rv[256];
    __shared__ int   ri[256];
    __shared__ int   sel[KNN_K];
    const int bm = blockIdx.x;          // b*128 + m
    const int b  = bm >> 7;
    const int t  = threadIdx.x;
    const float4* Pb = P4 + (size_t)b * NPTS;
    const float cx = cents[bm*4+0], cy = cents[bm*4+1];
    const float cz = cents[bm*4+2], cw = cents[bm*4+3];

    for (int i = t; i < NPTS; i += 256) {
        float4 q = Pb[i];
        float dx = q.x - cx, dy = q.y - cy, dz = q.z - cz, dw = q.w - cw;
        dd[i] = dx*dx + dy*dy + dz*dz + dw*dw;
    }
    __syncthreads();

    for (int k = 0; k < KNN_K; ++k) {
        float best = 3.4e38f; int bi = 0;
        for (int i = t; i < NPTS; i += 256) {
            float v = dd[i];
            if (v < best) { best = v; bi = i; }
        }
        rv[t] = best; ri[t] = bi;
        __syncthreads();
        for (int off = 128; off > 0; off >>= 1) {
            if (t < off) {
                float v2 = rv[t + off]; int i2 = ri[t + off];
                if (v2 < rv[t] || (v2 == rv[t] && i2 < ri[t])) { rv[t] = v2; ri[t] = i2; }
            }
            __syncthreads();
        }
        if (t == 0) { sel[k] = ri[0]; dd[ri[0]] = 3.4e38f; }
        __syncthreads();
    }

    for (int d = t; d < TOKDIM; d += 256) {
        float m = -3.4e38f;
#pragma unroll
        for (int k = 0; k < KNN_K; ++k) {
            float v = (float)feats[(size_t)(b * NPTS + sel[k]) * TOKDIM + d];
            m = fmaxf(m, v);
        }
        pooled[(size_t)bm * TOKDIM + d] = (__bf16)m;
    }
}

// ---------------------------------------------------------------------------
// Host-side orchestration
// ---------------------------------------------------------------------------
extern "C" void kernel_launch(void* const* d_in, const int* in_sizes, int n_in,
                              void* d_out, int out_size, void* d_ws, size_t ws_size,
                              hipStream_t stream) {
    (void)in_sizes; (void)n_in; (void)out_size; (void)ws_size;

    const float* coords   = (const float*)d_in[0];
    const float* features = (const float*)d_in[1];
    /* d_in[2] = batch_ids (int64): sorted, equal counts -> reshape semantics */
    const float* times    = (const float*)d_in[3];
    const float* Wm[4]    = {(const float*)d_in[4], (const float*)d_in[6],
                             (const float*)d_in[8], (const float*)d_in[10]};
    const float* bm[4]    = {(const float*)d_in[5], (const float*)d_in[7],
                             (const float*)d_in[9], (const float*)d_in[11]};
    const float* V1 = (const float*)d_in[12]; const float* c1 = (const float*)d_in[13];
    const float* V2 = (const float*)d_in[14]; const float* c2 = (const float*)d_in[15];

    // Output layout: tokens [8,128,768] f32 | centroids [8,128,4] | masks [8,128]
    float* tokens    = (float*)d_out;
    float* cents_out = tokens + (size_t)BATCH * MAXTOK * TOKDIM;      // +786432
    float* masks_out = cents_out + (size_t)BATCH * MAXTOK * 4;        // +790528

    // Workspace carve-up (~413 MB; 256B aligned chunks).
    char* ws = (char*)d_ws;
    size_t off = 0;
    auto carve = [&](size_t bytes) -> char* {
        char* p = ws + off;
        off += (bytes + 255) & ~(size_t)255;
        return p;
    };
    float4* P4     = (float4*)carve((size_t)NTOT * sizeof(float4));
    __bf16* actA   = (__bf16*)carve((size_t)NTOT * TOKDIM * 2);   // ping
    __bf16* actB   = (__bf16*)carve((size_t)NTOT * TOKDIM * 2);   // pong
    __bf16* W1t    = (__bf16*)carve((size_t)FEAT * 256 * 2);
    __bf16* W2t    = (__bf16*)carve((size_t)256 * 512 * 2);
    __bf16* W3t    = (__bf16*)carve((size_t)512 * 768 * 2);
    __bf16* W4t    = (__bf16*)carve((size_t)768 * 768 * 2);
    __bf16* V1t    = (__bf16*)carve((size_t)768 * 768 * 2);
    __bf16* V2t    = (__bf16*)carve((size_t)768 * 768 * 2);
    __bf16* pooled = (__bf16*)carve((size_t)BATCH * MAXTOK * TOKDIM * 2);
    __bf16* t1     = (__bf16*)carve((size_t)BATCH * MAXTOK * TOKDIM * 2);
    int*    fpsidx = (int*)   carve((size_t)BATCH * MAXTOK * 4);
    float*  centsW = (float*) carve((size_t)BATCH * MAXTOK * 4 * 4);

    const dim3 blk256(256);

    // Prep: points4, bf16 features, transposed bf16 weights.
    pack_points<<<(NTOT + 255) / 256, blk256, 0, stream>>>(coords, times, P4, NTOT);
    cvt_f32_bf16<<<(NTOT * FEAT + 255) / 256, blk256, 0, stream>>>(features, actA, NTOT * FEAT);
    cvt_transpose_w<<<(FEAT * 256 + 255) / 256, blk256, 0, stream>>>(Wm[0], W1t, FEAT, 256);
    cvt_transpose_w<<<(256 * 512 + 255) / 256, blk256, 0, stream>>>(Wm[1], W2t, 256, 512);
    cvt_transpose_w<<<(512 * 768 + 255) / 256, blk256, 0, stream>>>(Wm[2], W3t, 512, 768);
    cvt_transpose_w<<<(768 * 768 + 255) / 256, blk256, 0, stream>>>(Wm[3], W4t, 768, 768);
    cvt_transpose_w<<<(768 * 768 + 255) / 256, blk256, 0, stream>>>(V1, V1t, 768, 768);
    cvt_transpose_w<<<(768 * 768 + 255) / 256, blk256, 0, stream>>>(V2, V2t, 768, 768);

    // MLP1: 32 -> 256 -> 512 -> 768 -> 768 (GELU on first three, last linear).
    gemm_bf16_wmma<<<dim3(256/128, NTOT/128), blk256, 0, stream>>>(actA, W1t, bm[0], actB, NTOT, 256, 32,  1, 0);
    gemm_bf16_wmma<<<dim3(512/128, NTOT/128), blk256, 0, stream>>>(actB, W2t, bm[1], actA, NTOT, 512, 256, 1, 0);
    gemm_bf16_wmma<<<dim3(768/128, NTOT/128), blk256, 0, stream>>>(actA, W3t, bm[2], actB, NTOT, 768, 512, 1, 0);
    gemm_bf16_wmma<<<dim3(768/128, NTOT/128), blk256, 0, stream>>>(actB, W4t, bm[3], actA, NTOT, 768, 768, 0, 0);
    // feats (bf16) now in actA.

    // FPS centroids + kNN max-pool.
    fps_kernel<<<BATCH, 1024, 0, stream>>>(P4, fpsidx, centsW, cents_out);
    knn_pool_kernel<<<BATCH * MAXTOK, blk256, 0, stream>>>(P4, centsW, actA, pooled);

    // MLP2: 768 -> GELU -> 768, final layer writes f32 tokens directly to d_out.
    gemm_bf16_wmma<<<dim3(768/128, (BATCH*MAXTOK)/128), blk256, 0, stream>>>(pooled, V1t, c1, t1, BATCH*MAXTOK, 768, 768, 1, 0);
    gemm_bf16_wmma<<<dim3(768/128, (BATCH*MAXTOK)/128), blk256, 0, stream>>>(t1, V2t, c2, (void*)tokens, BATCH*MAXTOK, 768, 768, 0, 1);

    // Large-batch path: all tokens valid -> masks = 1 (mask multiply is identity).
    fill_ones<<<(BATCH * MAXTOK + 255) / 256, blk256, 0, stream>>>(masks_out, BATCH * MAXTOK);
}
